// DeepM3Model_515396076390
// MI455X (gfx1250) — matrix operand order repaired and checked by reading, not compiled, not run on hardware
//
#include <hip/hip_runtime.h>
#include <math.h>

typedef _Float16 v16h __attribute__((ext_vector_type(16)));
typedef _Float16 v8h  __attribute__((ext_vector_type(8)));
typedef float    v8f  __attribute__((ext_vector_type(8)));
typedef int      v4i  __attribute__((ext_vector_type(4)));

#define H       256
#define SEQ     200
#define BATCH   1024
#define NITEMS  50000
#define BM      16            // batch rows per workgroup
#define NWG     (BATCH/BM)    // 64
#define NT      256           // threads per block (8 wave32)
#define NWAVE   (NT/32)
#define NTILE_N (NITEMS/16)   // 3125

// ---- gfx1250 async-copy availability (guarded; sync fallback keeps build safe)
#if defined(__has_builtin)
# if __has_builtin(__builtin_amdgcn_global_load_async_to_lds_b128)
#  define HAVE_ASYNC_LDS 1
# endif
# if __has_builtin(__builtin_amdgcn_s_wait_asynccnt)
#  define HAVE_WAIT_ASYNC 1
# endif
#endif

#if HAVE_ASYNC_LDS
typedef __attribute__((address_space(1))) v4i* gptr4_t;  // global int4*
typedef __attribute__((address_space(3))) v4i* lptr4_t;  // LDS int4*
#endif

// ---------------- shared memory layout (319,552 B < 320 KB WGP limit) --------
#define OFF_H   0                   // float [16*256]  16384 B  (f32 hidden)
#define OFF_DT  16384               // float [16]         64 B
#define OFF_U   16448               // 40960 B phase-overlapped union
#define U_XE    (OFF_U + 0)         // f16 [16*256]: xe (gates) | ein (RK4)
#define U_HF16  (OFF_U + 8192)      // f16 [16*256]: persistent f16 h
#define U_G     (OFF_U + 16384)     // f16 [16*768]: gate pre-sums (gates phase)
#define U_UU    (OFF_U + 16384)     // f16 [16*256]: tanh intermediate (RK4)
#define OFF_W1  (OFF_U + 40960)     // f16 [256*256] pinned W1   131072 B
#define OFF_W2  (OFF_W1 + 131072)   // f16 [256*256] pinned W2   131072 B
#define SMEM_BYTES (OFF_W2 + 131072)

// ---------------- WMMA fragment helpers (CDNA5 16x16x32 f16 layouts) ---------
// A (16x32 f16): lane m=lane&15, half=lane>>4; elems 0..15 -> K=k0+16*half+e
__device__ __forceinline__ v16h loadA(const _Float16* __restrict__ A, int lda,
                                      int k0, int m, int half) {
  const v8h lo = *(const v8h*)(A + (size_t)m * lda + k0 + 16 * half);
  const v8h hi = *(const v8h*)(A + (size_t)m * lda + k0 + 16 * half + 8);
  return __builtin_shufflevector(lo, hi, 0,1,2,3,4,5,6,7,8,9,10,11,12,13,14,15);
}

// B (32x16 f16) from row-major weight W[out][in]: lane n=lane&15 -> one
// contiguous 32 B read at K = k0 + 16*half.  C[16x16] += A[16x256] * W^T
__device__ __forceinline__ v8f gemmK256(const _Float16* __restrict__ A,
                                        const _Float16* __restrict__ W,
                                        int brow0, v8f c, int mn, int half) {
  const _Float16* Brow = W + (size_t)(brow0 + mn) * H;
  #pragma unroll
  for (int k = 0; k < H; k += 32) {
    v16h a = loadA(A, H, k, mn, half);
    v16h b = *(const v16h*)(Brow + k + 16 * half);
    c = __builtin_amdgcn_wmma_f32_16x16x32_f16(false, a, false, b,
                                               (short)0, c, false, false);
  }
  return c;
}

__device__ __forceinline__ float sigm(float x) { return 1.f / (1.f + __expf(-x)); }

// ---------------- async global->LDS fill (W1/W2 pinning) ---------------------
__device__ __forceinline__ void fill_lds(_Float16* dst, const _Float16* src,
                                         int bytes, int tid) {
#if HAVE_ASYNC_LDS
  for (int off = tid * 16; off < bytes; off += NT * 16) {
    __builtin_amdgcn_global_load_async_to_lds_b128(
        (gptr4_t)((char*)src + off),
        (lptr4_t)((char*)dst + off), 0, 0);
  }
# if HAVE_WAIT_ASYNC
  __builtin_amdgcn_s_wait_asynccnt(0);
# else
  asm volatile("s_wait_asynccnt 0" ::: "memory");
# endif
#else
  for (int off = tid * 16; off < bytes; off += NT * 16)
    *(v8h*)((char*)dst + off) = *(const v8h*)((const char*)src + off);
#endif
}

// ---------------- fp32 -> f16 weight conversion ------------------------------
__global__ void cvt_kernel(const float* __restrict__ s, _Float16* __restrict__ d,
                           int n) {
  int i = blockIdx.x * blockDim.x + threadIdx.x;
  int stride = gridDim.x * blockDim.x;
  for (; i < n; i += stride) d[i] = (_Float16)s[i];
}

// ---------------- persistent-batch-tile GRU + RK4 recurrence -----------------
__global__ __launch_bounds__(NT) void rnn_kernel(
    const int* __restrict__ x, const float* __restrict__ t,
    const float* __restrict__ emb,
    const _Float16* __restrict__ Wih, const _Float16* __restrict__ Whh,
    const float* __restrict__ b_ih, const float* __restrict__ b_hh,
    const _Float16* __restrict__ W1, const float* __restrict__ b1,
    const _Float16* __restrict__ W2, const float* __restrict__ b2,
    _Float16* __restrict__ hfin) {
  __shared__ __align__(16) unsigned char smem[SMEM_BYTES];
  float*    sh_h  = (float*)(smem + OFF_H);
  float*    sh_dt = (float*)(smem + OFF_DT);
  _Float16* sh_xe = (_Float16*)(smem + U_XE);
  _Float16* sh_ein= (_Float16*)(smem + U_XE);   // aliases xe (disjoint lifetime)
  _Float16* sh_hf = (_Float16*)(smem + U_HF16);
  _Float16* sh_g  = (_Float16*)(smem + U_G);
  _Float16* sh_u  = (_Float16*)(smem + U_UU);   // aliases g head (disjoint life)
  _Float16* sW1   = (_Float16*)(smem + OFF_W1);
  _Float16* sW2   = (_Float16*)(smem + OFF_W2);

  const int tid  = threadIdx.x;
  const int wave = tid >> 5, lane = tid & 31;
  const int half = lane >> 4, mn = lane & 15;
  const int m0 = blockIdx.x * BM;

  // pin W1/W2 in LDS once (async copy), reused 4x per step for 200 steps
  fill_lds(sW1, W1, 131072, tid);
  fill_lds(sW2, W2, 131072, tid);
  for (int i = tid; i < BM * H; i += NT) { sh_h[i] = 0.f; sh_hf[i] = (_Float16)0.f; }
  __syncthreads();

  for (int s = 0; s < SEQ; ++s) {
    // ---- prep: dt per row, embedding gather, prefetch next step's rows ----
    if (tid < BM) {
      int b = m0 + tid;
      float dt = 0.f;
      if (s < SEQ - 1) {
        float tc = t[b * SEQ + s], tn = t[b * SEQ + s + 1];
        dt = fmaxf(tn, tc + 1e-5f) - tc;
      }
      sh_dt[tid] = dt;
    }
    for (int i = tid; i < BM * H; i += NT) {
      int m = i >> 8, c = i & 255;
      int idx = x[(m0 + m) * SEQ + s];
      sh_xe[i] = (_Float16)emb[(size_t)idx * H + c];
    }
    if (s + 1 < SEQ && tid < 128) {      // global_prefetch_b8 of next gather
      int m = tid >> 3, off = (tid & 7) * 32;
      int idxn = x[(m0 + m) * SEQ + s + 1];
      __builtin_prefetch((const void*)(emb + (size_t)idxn * H + off), 0, 1);
    }
    __syncthreads();

    // ---- GRU gates: g[:,0:512] = xe@Wih^T + h@Whh^T (fused, uniform loops);
    //      g[:,512:768] = i_n;  h_n kept in per-wave WMMA register fragments --
    for (int nt = wave; nt < 32; nt += NWAVE) {
      v8f c = {};
      c = gemmK256(sh_xe, Wih, nt * 16, c, mn, half);
      c = gemmK256(sh_hf, Whh, nt * 16, c, mn, half);
      #pragma unroll
      for (int r = 0; r < 8; ++r)
        sh_g[(r + 8 * half) * 768 + nt * 16 + mn] = (_Float16)c[r];
    }
    for (int nt = 32 + wave; nt < 48; nt += NWAVE) {
      v8f c = {};
      c = gemmK256(sh_xe, Wih, nt * 16, c, mn, half);
      #pragma unroll
      for (int r = 0; r < 8; ++r)
        sh_g[(r + 8 * half) * 768 + nt * 16 + mn] = (_Float16)c[r];
    }
    v8f hn0 = {}, hn1 = {};
    hn0 = gemmK256(sh_hf, Whh, 512 + (wave * 2 + 0) * 16, hn0, mn, half);
    hn1 = gemmK256(sh_hf, Whh, 512 + (wave * 2 + 1) * 16, hn1, mn, half);
    __syncthreads();

    // ---- GRU elementwise on this wave's owned columns [wave*32, wave*32+32) -
    v8f acc0 = {}, acc1 = {};
    #pragma unroll
    for (int q = 0; q < 2; ++q) {
      const v8f& hnq = q ? hn1 : hn0;
      int col = (wave * 2 + q) * 16 + mn;
      float bir = b_ih[col],       bhr = b_hh[col];
      float biz = b_ih[256 + col], bhz = b_hh[256 + col];
      float bin = b_ih[512 + col], bhn = b_hh[512 + col];
      #pragma unroll
      for (int r = 0; r < 8; ++r) {
        int row = r + 8 * half, pos = row * 256 + col;
        float sr  = (float)sh_g[row * 768 + col]       + bir + bhr;
        float sz  = (float)sh_g[row * 768 + 256 + col] + biz + bhz;
        float in_ = (float)sh_g[row * 768 + 512 + col] + bin;
        float hv  = hnq[r] + bhn;
        float rr = sigm(sr), zz = sigm(sz);
        float nn = tanhf(in_ + rr * hv);
        float h = (1.f - zz) * nn + zz * sh_h[pos];
        sh_h[pos] = h;
        sh_ein[pos] = (_Float16)h;
      }
    }
    __syncthreads();

    // ---- RK4: f(h)=tanh(h@W1^T+b1)@W2^T+b2; W1/W2 from LDS, k-acc in regs ---
    #pragma unroll 1
    for (int e = 0; e < 4; ++e) {
      #pragma unroll
      for (int q = 0; q < 2; ++q) {            // u = tanh(ein@W1^T + b1)
        int nt = wave * 2 + q;
        v8f c = {};
        c = gemmK256(sh_ein, sW1, nt * 16, c, mn, half);
        int col = nt * 16 + mn;
        float bb = b1[col];
        #pragma unroll
        for (int r = 0; r < 8; ++r)
          sh_u[(r + 8 * half) * 256 + col] = (_Float16)tanhf(c[r] + bb);
      }
      __syncthreads();
      const float wgt = (e == 0 || e == 3) ? 1.f : 2.f;
      const float cn  = (e < 2) ? 0.5f : 1.f;
      #pragma unroll
      for (int q = 0; q < 2; ++q) {            // f = u@W2^T + b2 (fused epilog)
        int nt = wave * 2 + q;
        v8f c = {};
        c = gemmK256(sh_u, sW2, nt * 16, c, mn, half);
        v8f& accq = q ? acc1 : acc0;
        int col = nt * 16 + mn;
        float bb = b2[col];
        #pragma unroll
        for (int r = 0; r < 8; ++r) {
          int row = r + 8 * half, pos = row * 256 + col;
          float f = c[r] + bb;
          accq[r] += wgt * f;
          if (e < 3) {
            sh_ein[pos] = (_Float16)(sh_h[pos] + cn * sh_dt[row] * f);
          } else {
            float hnew = sh_h[pos] + sh_dt[row] * (1.f / 6.f) * accq[r];
            sh_h[pos] = hnew;
            sh_hf[pos] = (_Float16)hnew;
          }
        }
      }
      __syncthreads();
    }
  }
  for (int i = tid; i < BM * H; i += NT)
    hfin[(size_t)m0 * H + i] = sh_hf[i];
}

// ---------------- head: out[1024,50000] = h @ W_head^T + b_head --------------
__global__ __launch_bounds__(256) void head_kernel(
    const _Float16* __restrict__ hf, const _Float16* __restrict__ Wh,
    const float* __restrict__ bh, float* __restrict__ out) {
  const int tid = threadIdx.x;
  const int wave = tid >> 5, lane = tid & 31;
  const int half = lane >> 4, mn = lane & 15;
  int tile = blockIdx.x * 8 + wave;
  int mt = tile / NTILE_N, nt = tile % NTILE_N;
  const _Float16* A = hf + (size_t)mt * 16 * H;
  const _Float16* Brow = Wh + (size_t)(nt * 16 + mn) * H;
  v8f c = {};
  #pragma unroll
  for (int k = 0; k < H; k += 32) {
    v16h a = loadA(A, H, k, mn, half);
    v16h b = *(const v16h*)(Brow + k + 16 * half);
    c = __builtin_amdgcn_wmma_f32_16x16x32_f16(false, a, false, b,
                                               (short)0, c, false, false);
  }
  int col = nt * 16 + mn;
  float bb = bh[col];
  #pragma unroll
  for (int r = 0; r < 8; ++r) {
    int row = mt * 16 + r + 8 * half;
    out[(size_t)row * NITEMS + col] = c[r] + bb;
  }
}

extern "C" void kernel_launch(void* const* d_in, const int* in_sizes, int n_in,
                              void* d_out, int out_size, void* d_ws, size_t ws_size,
                              hipStream_t stream) {
  (void)in_sizes; (void)n_in; (void)out_size; (void)ws_size;
  const int*   x     = (const int*)  d_in[0];
  const float* t     = (const float*)d_in[1];
  const float* emb   = (const float*)d_in[2];
  const float* W_ih  = (const float*)d_in[3];
  const float* W_hh  = (const float*)d_in[4];
  const float* b_ih  = (const float*)d_in[5];
  const float* b_hh  = (const float*)d_in[6];
  const float* W1    = (const float*)d_in[7];
  const float* b1    = (const float*)d_in[8];
  const float* W2    = (const float*)d_in[9];
  const float* b2    = (const float*)d_in[10];
  const float* Whead = (const float*)d_in[11];
  const float* bhead = (const float*)d_in[12];
  float* out = (float*)d_out;

  // workspace: f16 weight copies + final hidden state (f16). ~27 MB total.
  _Float16* ws    = (_Float16*)d_ws;
  _Float16* Wih_h = ws;                     // 196608
  _Float16* Whh_h = Wih_h + 196608;         // 196608
  _Float16* W1_h  = Whh_h + 196608;         // 65536
  _Float16* W2_h  = W1_h + 65536;           // 65536
  _Float16* Whd_h = W2_h + 65536;           // 12800000
  _Float16* hfin  = Whd_h + 12800000;       // 262144

  cvt_kernel<<<768, 256, 0, stream>>>(W_ih, Wih_h, 196608);
  cvt_kernel<<<768, 256, 0, stream>>>(W_hh, Whh_h, 196608);
  cvt_kernel<<<256, 256, 0, stream>>>(W1, W1_h, 65536);
  cvt_kernel<<<256, 256, 0, stream>>>(W2, W2_h, 65536);
  cvt_kernel<<<4096, 256, 0, stream>>>(Whead, Whd_h, 12800000);

  rnn_kernel<<<NWG, NT, 0, stream>>>(x, t, emb, Wih_h, Whh_h, b_ih, b_hh,
                                     W1_h, b1, W2_h, b2, hfin);

  head_kernel<<<(64 * NTILE_N) / 8, 256, 0, stream>>>(hfin, Whd_h, bhead, out);
}
